// GraphPool_55456617725995
// MI455X (gfx1250) — matrix operand order repaired and checked by reading, not compiled, not run on hardware
//
#include <hip/hip_runtime.h>
#include <hip/hip_bf16.h>
#include <stdint.h>

// ---------------------------------------------------------------------------
// GCN (2x GCNConv + MLP head) for MI455X / gfx1250, wave32, bf16 WMMA.
//  - GEMMs via v_wmma_f32_16x16x32_bf16, B fragments staged in LDS
//  - self-loop init fused into GEMM epilogue
//  - message passing via global_atomic_add_f32 (h and acc are L2-resident)
// ---------------------------------------------------------------------------

#define N_NODES 100000
#define N_EDGES 3200000
#define IN_F    384
#define HID     64
#define NC      2

typedef __attribute__((ext_vector_type(16))) __bf16 v16bf;
typedef __attribute__((ext_vector_type(8)))  float  v8f;

// ---- scalar bf16 RNE pack (used by weight-prep kernel) --------------------
__device__ __forceinline__ unsigned short f2bf(float f) {
    unsigned u = __float_as_uint(f);
    u += 0x7FFFu + ((u >> 16) & 1u);
    return (unsigned short)(u >> 16);
}

// ---------------------------------------------------------------------------
// Pack a K x 64 f32 weight matrix into WMMA B-fragment order (bf16).
// B 16-bit layout (32x16 tile): lane n (0..15) holds K=0..7,16..23,
// lane n+16 holds K=8..15,24..31. frag index:
//   ((kb*4 + c)*32 + lane)*16 + pos ,
//   K = kb*32 + (pos>>3)*16 + (lane>>4)*8 + (pos&7),
//   col = c*16 + (lane&15)
// ---------------------------------------------------------------------------
__global__ void prep_wfrag_kernel(const float* __restrict__ W, int K,
                                  unsigned short* __restrict__ frag) {
    int total = K * 64;
    for (int t = blockIdx.x * blockDim.x + threadIdx.x; t < total;
         t += gridDim.x * blockDim.x) {
        int pos  = t & 15;
        int lane = (t >> 4) & 31;
        int c    = (t >> 9) & 3;
        int kb   = t >> 11;
        int kidx = kb * 32 + (pos >> 3) * 16 + (lane >> 4) * 8 + (pos & 7);
        int col  = c * 16 + (lane & 15);
        frag[t] = f2bf(W[kidx * 64 + col]);
    }
}

// ---- degree / normalization ----------------------------------------------
__global__ void init_deg_kernel(float* __restrict__ deg) {
    int i = blockIdx.x * blockDim.x + threadIdx.x;
    if (i < N_NODES) deg[i] = 1.0f;                  // self-loop weight
}
__global__ void edge_deg_kernel(const long long* __restrict__ ei,
                                const float* __restrict__ ew,
                                float* __restrict__ deg) {
    int e = blockIdx.x * blockDim.x + threadIdx.x;
    if (e < N_EDGES) {
        long long c = ei[(long long)N_EDGES + e];
        atomicAdd(&deg[(int)c], ew[e]);
    }
}
__global__ void deg_to_dis_kernel(float* __restrict__ deg) {
    int i = blockIdx.x * blockDim.x + threadIdx.x;
    if (i < N_NODES) {
        float d = deg[i];
        deg[i] = (d > 0.0f) ? rsqrtf(d) : 0.0f;
    }
}

// ---------------------------------------------------------------------------
// WMMA GEMM:  C[nrows x 64] = act(A[nrows x K]) @ W[K x 64]
// Optional fused outputs:
//   ELU_BIAS_OUT : C = ELU(C + bias)
//   SELF_OUT     : also write Cacc = dis[row]^2 * C + bias2[col]
// block = 256 threads = 8 waves; each wave: 16 rows x 64 cols (4 wmma tiles).
// B fragments staged once per block into LDS.
// ---------------------------------------------------------------------------
template <int K, bool LRELU_IN, bool ELU_BIAS_OUT, bool SELF_OUT>
__global__ __launch_bounds__(256) void gemm_wmma_kernel(
    const float* __restrict__ A, const unsigned short* __restrict__ Bfrag,
    const float* __restrict__ bias, const float* __restrict__ dis,
    float* __restrict__ C, float* __restrict__ Cacc, int nrows) {
    __shared__ unsigned short sB[K * 64];

    // cooperative copy of B fragments (K*64 bf16 = K*128 bytes) into LDS
    {
        const uint4* src = (const uint4*)Bfrag;
        uint4* dst = (uint4*)sB;
        const int n16 = (K * 64) / 8;              // uint4 count
#pragma unroll
        for (int i = threadIdx.x; i < n16; i += 256) dst[i] = src[i];
    }
    __syncthreads();

    const int lane = threadIdx.x & 31;
    const int wave = threadIdx.x >> 5;
    const int g    = lane >> 4;    // lane group (0/1)
    const int ln   = lane & 15;
    const int rowBase = blockIdx.x * 128 + wave * 16;
    const int aRow    = rowBase + ln;
    const bool valid  = aRow < nrows;
    const size_t aOff = (size_t)aRow * K;

    v8f acc[4];
#pragma unroll
    for (int c = 0; c < 4; ++c) acc[c] = (v8f){0.f,0.f,0.f,0.f,0.f,0.f,0.f,0.f};

    union AF { v16bf bf; unsigned int u[8]; } a;
    union BF { v16bf bf; uint4 q[2]; } b;

#pragma unroll
    for (int kb = 0; kb < K / 32; ++kb) {
        // A-fragment: lane group g holds K = {kb*32+g*8..+7} U {kb*32+16+g*8..+7}
        float f[16];
        if (valid) {
            const float4* p0 = (const float4*)(A + aOff + kb * 32 + g * 8);
            const float4* p1 = (const float4*)(A + aOff + kb * 32 + 16 + g * 8);
            float4 x0 = p0[0], x1 = p0[1], x2 = p1[0], x3 = p1[1];
            f[0]=x0.x; f[1]=x0.y; f[2]=x0.z;  f[3]=x0.w;
            f[4]=x1.x; f[5]=x1.y; f[6]=x1.z;  f[7]=x1.w;
            f[8]=x2.x; f[9]=x2.y; f[10]=x2.z; f[11]=x2.w;
            f[12]=x3.x;f[13]=x3.y;f[14]=x3.z; f[15]=x3.w;
        } else {
#pragma unroll
            for (int j = 0; j < 16; ++j) f[j] = 0.0f;
        }
        if (LRELU_IN) {
#pragma unroll
            for (int j = 0; j < 16; ++j) f[j] = f[j] > 0.0f ? f[j] : 0.01f * f[j];
        }
        // native f32 -> bf16 conversion (RNE)
#pragma unroll
        for (int j = 0; j < 16; ++j) a.bf[j] = (__bf16)f[j];

#pragma unroll
        for (int c = 0; c < 4; ++c) {
            const uint4* bp =
                (const uint4*)(sB + ((((kb * 4 + c) * 32) + lane) << 4));
            b.q[0] = bp[0];
            b.q[1] = bp[1];
            acc[c] = __builtin_amdgcn_wmma_f32_16x16x32_bf16(
                false, a.bf, false, b.bf, (short)0, acc[c], false, false);
        }
    }

    // epilogue: vgpr r -> M = r + g*8 ; col = c*16 + ln
    float bv[4];
#pragma unroll
    for (int c = 0; c < 4; ++c)
        bv[c] = (ELU_BIAS_OUT || SELF_OUT) ? bias[c * 16 + ln] : 0.0f;

#pragma unroll
    for (int r = 0; r < 8; ++r) {
        const int row = rowBase + g * 8 + r;
        const bool inb = row < nrows;
        float d2 = 0.0f;
        if (SELF_OUT && inb) {
            float dv = dis[row];
            d2 = dv * dv;
        }
#pragma unroll
        for (int c = 0; c < 4; ++c) {
            const int col = c * 16 + ln;
            float v = acc[c][r];
            if (ELU_BIAS_OUT) {
                v += bv[c];
                v = v > 0.0f ? v : (__expf(v) - 1.0f);   // ELU(alpha=1)
            }
            if (inb) {
                C[(size_t)row * 64 + col] = v;
                if (SELF_OUT)
                    Cacc[(size_t)row * 64 + col] = d2 * v + bv[c];
            }
        }
    }
}

// ---- edge scatter: one wave per edge, 2 features per lane -----------------
__global__ void edge_scatter_kernel(const long long* __restrict__ ei,
                                    const float* __restrict__ ew,
                                    const float* __restrict__ dis,
                                    const float* __restrict__ h,
                                    float* __restrict__ acc) {
    long long t = (long long)blockIdx.x * blockDim.x + threadIdx.x;
    long long e = t >> 5;
    int lane = (int)(t & 31);
    if (e < N_EDGES) {
        int r = (int)ei[e];
        int c = (int)ei[(long long)N_EDGES + e];
        float nrm = dis[r] * ew[e] * dis[c];
        size_t ro = (size_t)r * 64, co = (size_t)c * 64;
        atomicAdd(&acc[co + lane],      nrm * h[ro + lane]);
        atomicAdd(&acc[co + lane + 32], nrm * h[ro + lane + 32]);
    }
}

// ---- MLP head: logits = h @ Wm2 + bm2 ; softmax over 2 classes ------------
__global__ void mlp2_softmax_kernel(const float* __restrict__ h,
                                    const float* __restrict__ Wm2,
                                    const float* __restrict__ bm2,
                                    float* __restrict__ out) {
    int i = blockIdx.x * blockDim.x + threadIdx.x;
    if (i < N_NODES) {
        const float* hp = h + (size_t)i * 64;
        float l0 = bm2[0], l1 = bm2[1];
#pragma unroll 8
        for (int j = 0; j < 64; ++j) {
            float v = hp[j];
            l0 += v * Wm2[j * 2 + 0];
            l1 += v * Wm2[j * 2 + 1];
        }
        float m  = l0 > l1 ? l0 : l1;
        float e0 = __expf(l0 - m), e1 = __expf(l1 - m);
        float s  = 1.0f / (e0 + e1);
        out[2 * i + 0] = e0 * s;
        out[2 * i + 1] = e1 * s;
    }
}

// ---------------------------------------------------------------------------
// workspace layout (bytes, 256-aligned)
// ---------------------------------------------------------------------------
constexpr size_t alup(size_t x) { return (x + 255) & ~(size_t)255; }
constexpr size_t OFF_DIS  = 0;
constexpr size_t OFF_W1F  = OFF_DIS  + alup((size_t)N_NODES * 4);
constexpr size_t OFF_W2F  = OFF_W1F  + alup((size_t)IN_F * HID * 2);
constexpr size_t OFF_WM1F = OFF_W2F  + alup((size_t)HID * HID * 2);
constexpr size_t OFF_B0   = OFF_WM1F + alup((size_t)HID * HID * 2);
constexpr size_t SZ_BUF   = alup((size_t)N_NODES * HID * 4);
constexpr size_t OFF_B1   = OFF_B0 + SZ_BUF;
constexpr size_t OFF_B2   = OFF_B1 + SZ_BUF;

extern "C" void kernel_launch(void* const* d_in, const int* in_sizes, int n_in,
                              void* d_out, int out_size, void* d_ws,
                              size_t ws_size, hipStream_t stream) {
    const float*     X   = (const float*)d_in[0];
    const long long* ei  = (const long long*)d_in[1];   // int64 (2, E)
    const float*     ew  = (const float*)d_in[2];
    const float*     W1  = (const float*)d_in[3];
    const float*     b1  = (const float*)d_in[4];
    const float*     W2  = (const float*)d_in[5];
    const float*     b2  = (const float*)d_in[6];
    const float*     Wm1 = (const float*)d_in[7];
    const float*     bm1 = (const float*)d_in[8];
    const float*     Wm2 = (const float*)d_in[9];
    const float*     bm2 = (const float*)d_in[10];
    float* out = (float*)d_out;

    char* ws = (char*)d_ws;
    float*          dis  = (float*)(ws + OFF_DIS);
    unsigned short* w1f  = (unsigned short*)(ws + OFF_W1F);
    unsigned short* w2f  = (unsigned short*)(ws + OFF_W2F);
    unsigned short* wm1f = (unsigned short*)(ws + OFF_WM1F);
    float* B0 = (float*)(ws + OFF_B0);
    float* B1 = (float*)(ws + OFF_B1);
    float* B2 = (float*)(ws + OFF_B2);

    const int T = 256;
    const int gemm_blocks = (N_NODES + 127) / 128;
    const int scat_blocks = (int)(((long long)N_EDGES * 32 + T - 1) / T);

    // 1) pack weights to bf16 WMMA B-fragments
    prep_wfrag_kernel<<<(IN_F * 64 + T - 1) / T, T, 0, stream>>>(W1, IN_F, w1f);
    prep_wfrag_kernel<<<(HID * 64 + T - 1) / T, T, 0, stream>>>(W2, HID, w2f);
    prep_wfrag_kernel<<<(HID * 64 + T - 1) / T, T, 0, stream>>>(Wm1, HID, wm1f);

    // 2) gcn_norm: deg (with self-loops) -> dis = rsqrt(deg)
    init_deg_kernel<<<(N_NODES + T - 1) / T, T, 0, stream>>>(dis);
    edge_deg_kernel<<<(N_EDGES + T - 1) / T, T, 0, stream>>>(ei, ew, dis);
    deg_to_dis_kernel<<<(N_NODES + T - 1) / T, T, 0, stream>>>(dis);

    // 3) conv1: h1 = X @ W1 ; acc1 = dis^2*h1 + b1 (fused)
    gemm_wmma_kernel<IN_F, false, false, true><<<gemm_blocks, T, 0, stream>>>(
        X, w1f, b1, dis, B0, B1, N_NODES);
    edge_scatter_kernel<<<scat_blocks, T, 0, stream>>>(ei, ew, dis, B0, B1);

    // 4) conv2: h2 = lrelu(acc1) @ W2 ; acc2 = dis^2*h2 + b2 (fused)
    gemm_wmma_kernel<HID, true, false, true><<<gemm_blocks, T, 0, stream>>>(
        B1, w2f, b2, dis, B2, B0, N_NODES);
    edge_scatter_kernel<<<scat_blocks, T, 0, stream>>>(ei, ew, dis, B2, B0);

    // 5) MLP layer 1: h3 = ELU(acc2 @ Wm1 + bm1)
    gemm_wmma_kernel<HID, false, true, false><<<gemm_blocks, T, 0, stream>>>(
        B0, wm1f, bm1, nullptr, B1, nullptr, N_NODES);

    // 6) MLP layer 2 + softmax -> out [N, 2]
    mlp2_softmax_kernel<<<(N_NODES + T - 1) / T, T, 0, stream>>>(B1, Wm2, bm2, out);

    (void)in_sizes; (void)n_in; (void)out_size; (void)ws_size;
}
